// MambaFourierSSM_13091060318313
// MI455X (gfx1250) — matrix operand order repaired and checked by reading, not compiled, not run on hardware
//
#include <hip/hip_runtime.h>
#include <cstdint>
#include <cstddef>

// ---------------------------------------------------------------------------
// MambaFourierSSM for MI455X (gfx1250, wave32, WMMA + Tensor Data Mover)
//
// Pass 0a: convert weights f32 -> f16, pre-transposed Wt[g][n][k] (768 KB).
// Pass 0b: convert x_re/x_im -> f16 xh[p][512] (re||im, 64 MB, L2-resident).
// Pass 0c: separable DFT row transform G[g][i][w][c] (5.5 MB, L2-resident):
//          G = sum_j K_g[c,i,j] * cis(-2*pi*w*(60+j)/128).
// Pass 1 : gate GEMM (65536x512)@(512x256) x3 via f16 WMMA 16x16x32 (f32 acc);
//          all LDS staging by TENSOR_LOAD_TO_LDS (double-buffered,
//          s_wait_tensorcnt); fused sigmoid/softplus epilogue; packed f16
//          {fg, delta*ig} gates to workspace.
// Pass 2 : one thread per (h,w,c): fields from 7 complex FMAs against G
//          (h-twiddles via 1 sincos pair + recurrence), then loop over the
//          4 batch elements; closed-form scan sum (1+d)(1+d^2)(1+d^4).
// ---------------------------------------------------------------------------

typedef __attribute__((ext_vector_type(16))) _Float16 v16h;
typedef __attribute__((ext_vector_type(8)))  _Float16 v8h;
typedef __attribute__((ext_vector_type(8)))  float    v8f;
typedef __attribute__((ext_vector_type(4)))  unsigned int u32x4;
typedef __attribute__((ext_vector_type(8)))  int i32x8;
typedef __attribute__((ext_vector_type(4)))  int i32x4;

#define BB   4
#define HH   128
#define WWD  128
#define CC   256
#define TWOC 512
#define PIX  (BB * HH * WWD)        // 65536 pixels
#define HWCN (HH * WWD * CC)        // 4,194,304 per batch

// Toolchain arity probe: the TDM header ships only with the 6-arg toolchain.
#if __has_include(<hip/amd_detail/amd_gfx1250_TDM.h>)
#define TDM_LOAD(g0, g1, g2, g3) \
  __builtin_amdgcn_tensor_load_to_lds((g0), (g1), (g2), (g3), (i32x8){0,0,0,0,0,0,0,0}, 0)
#else
#define TDM_LOAD(g0, g1, g2, g3) \
  __builtin_amdgcn_tensor_load_to_lds((g0), (g1), (g2), (g3), 0)
#endif

__device__ __forceinline__ float2 cmulf(float2 a, float2 b) {
  return make_float2(a.x * b.x - a.y * b.y, a.x * b.y + a.y * b.x);
}

// Issue one 2-D TDM tile load: f16 elements, tile_d0 x tile_d1, row stride
// stride0 (elements), LDS padding of 4 DWORDs every 16 DWORDs stored
// (row = 32 halves = 16 DW) -> LDS row stride 40 halves.
__device__ __forceinline__ void tdm_issue(uint32_t lds_off, uint64_t gaddr,
                                          uint32_t tile_d0, uint32_t tile_d1,
                                          uint32_t tens_d0, uint32_t tens_d1,
                                          uint32_t stride0) {
  u32x4 g0;
  g0[0] = 1u;                                        // count=1, user mode
  g0[1] = lds_off;                                   // lds_addr (bytes)
  g0[2] = (uint32_t)(gaddr & 0xffffffffu);           // global_addr[31:0]
  g0[3] = (uint32_t)((gaddr >> 32) & 0x1ffffffu)     // global_addr[56:32]
          | (2u << 30);                               // type = 2 ("image")
  i32x8 g1;
  g1[0] = (int)((1u << 16)                           // data_size = 2 bytes
              | (1u << 20)                           // pad_enable
              | (3u << 22)                           // pad_interval: 16 DW
              | (3u << 25));                         // pad_amount: 4 DW
  g1[1] = (int)((tens_d0 & 0xffffu) << 16);          // tensor_dim0[15:0]
  g1[2] = (int)((tens_d0 >> 16) | ((tens_d1 & 0xffffu) << 16));
  g1[3] = (int)((tens_d1 >> 16) | (tile_d0 << 16));  // tile_dim0
  g1[4] = (int)(tile_d1 & 0xffffu);                  // tile_dim1 (tile_dim2=0)
  g1[5] = (int)stride0;                              // tensor_dim0_stride[31:0]
  g1[6] = 0;
  g1[7] = 0;
  i32x4 z4 = {0, 0, 0, 0};
  TDM_LOAD(g0, g1, z4, z4);
}

// ---------------------------------------------------------------------------
// Pass 0a: weights f32 (512,256) -> f16 transposed (256,512), x3 gates.
// ---------------------------------------------------------------------------
__global__ __launch_bounds__(256) void convert_w(
    const float* __restrict__ Wi, const float* __restrict__ Wf,
    const float* __restrict__ Wd, _Float16* __restrict__ Wt)
{
  const unsigned idx = blockIdx.x * 256u + threadIdx.x;   // < 3*16384
  const unsigned g  = idx >> 14;
  const unsigned r  = idx & 16383u;
  const unsigned n  = r >> 6;
  const unsigned k8 = (r & 63u) * 8u;
  const float* W = (g == 0) ? Wi : (g == 1) ? Wf : Wd;
  v8h h;
#pragma unroll
  for (int u = 0; u < 8; ++u)
    h[u] = (_Float16)W[(size_t)(k8 + u) * CC + n];
  *(v8h*)(Wt + (size_t)g * (CC * TWOC) + (size_t)n * TWOC + k8) = h;
}

// ---------------------------------------------------------------------------
// Pass 0b: x_re/x_im f32 -> xh f16 [p][512] (re || im).
// ---------------------------------------------------------------------------
__global__ __launch_bounds__(256) void convert_x(
    const float* __restrict__ xre, const float* __restrict__ xim,
    _Float16* __restrict__ xh)
{
  const unsigned idx = blockIdx.x * 256u + threadIdx.x;   // < PIX*32
  const unsigned p  = idx >> 5;
  const unsigned c8 = (idx & 31u) * 8u;
  const float* r = xre + (size_t)p * CC + c8;
  const float* m = xim + (size_t)p * CC + c8;
  float4 r0 = *(const float4*)r, r1 = *(const float4*)(r + 4);
  float4 m0 = *(const float4*)m, m1 = *(const float4*)(m + 4);
  v8h hr, hm;
  hr[0] = (_Float16)r0.x; hr[1] = (_Float16)r0.y; hr[2] = (_Float16)r0.z; hr[3] = (_Float16)r0.w;
  hr[4] = (_Float16)r1.x; hr[5] = (_Float16)r1.y; hr[6] = (_Float16)r1.z; hr[7] = (_Float16)r1.w;
  hm[0] = (_Float16)m0.x; hm[1] = (_Float16)m0.y; hm[2] = (_Float16)m0.z; hm[3] = (_Float16)m0.w;
  hm[4] = (_Float16)m1.x; hm[5] = (_Float16)m1.y; hm[6] = (_Float16)m1.z; hm[7] = (_Float16)m1.w;
  *(v8h*)(xh + (size_t)p * TWOC + c8) = hr;
  *(v8h*)(xh + (size_t)p * TWOC + CC + c8) = hm;
}

// ---------------------------------------------------------------------------
// Pass 0c: DFT row transform. idx enumerates (g, i, w, c) with c fastest,
// so out index == idx (layout G[g][i][w][c], complex interleaved).
// ---------------------------------------------------------------------------
__global__ __launch_bounds__(256) void build_rowfft(
    const float* __restrict__ Ak, const float* __restrict__ Bk,
    const float* __restrict__ Ck, float2* __restrict__ G)
{
  const unsigned idx = blockIdx.x * 256u + threadIdx.x;   // < 3*7*128*256
  const unsigned c  = idx & 255u;
  const unsigned w  = (idx >> 8) & 127u;
  const unsigned gi = idx >> 15;     // 0..20
  const unsigned g  = gi / 7u;
  const unsigned i  = gi % 7u;
  const float* K = (g == 0) ? Ak : (g == 1) ? Bk : Ck;
  const float* kr = K + c * 49 + i * 7;

  const float STEP = -0.04908738521234052f;  // -2*pi/128
  float2 acc = make_float2(0.f, 0.f);
#pragma unroll
  for (int j = 0; j < 7; ++j) {
    const int pw = (int)((w * (60u + j)) & 127u);
    float sj, cj;
    __sincosf((float)pw * STEP, &sj, &cj);
    const float kv = kr[j];
    acc.x = fmaf(kv, cj, acc.x);
    acc.y = fmaf(kv, sj, acc.y);
  }
  G[idx] = acc;
}

// ---------------------------------------------------------------------------
// Pass 1: gates GEMM. Block: 256 threads (8 waves). Tile M=64 x N=128, K+=32.
// Wave (wm 0..3, wn 0..1): 4 WMMA tiles x 3 gates. grid = (PIX/64, 2).
// Staging by TDM, double-buffered.
// ---------------------------------------------------------------------------
__global__ __launch_bounds__(256) void gates_gemm_wmma(
    const _Float16* __restrict__ xh, const _Float16* __restrict__ Wt,
    const float* __restrict__ bi, const float* __restrict__ bf,
    const float* __restrict__ bd,
    unsigned int* __restrict__ gates)
{
  __shared__ _Float16 Alds[2][64][40];        // [buf][m][k], stride 40 halves
  __shared__ _Float16 Blds[2][3][128][40];    // [buf][gate][n][k] (transposed)

  const int tid   = threadIdx.x;
  const int lane  = tid & 31;
  const int wv    = tid >> 5;
  const int wm    = wv >> 1;          // 0..3
  const int wn    = wv & 1;           // 0..1
  const int pBase = blockIdx.x * 64;
  const int nBase = blockIdx.y * 128;
  const int ml    = lane & 15;
  const int hi    = lane >> 4;

  const uint32_t aldsOff = (uint32_t)(uintptr_t)(&Alds[0][0][0]);
  const uint32_t bldsOff = (uint32_t)(uintptr_t)(&Blds[0][0][0][0]);
  const uint64_t xhAddr  = (uint64_t)(uintptr_t)xh;
  const uint64_t wtAddr  = (uint64_t)(uintptr_t)Wt;

  v8f acc[3][4];
#pragma unroll
  for (int g = 0; g < 3; ++g)
#pragma unroll
    for (int t = 0; t < 4; ++t)
#pragma unroll
      for (int e = 0; e < 8; ++e) acc[g][t][e] = 0.0f;

  // one wave issues the DMA batch (EXEC is ignored by TDM; 1 wave = 1 DMA)
  auto issue = [&](int buf, int k) {
    tdm_issue(aldsOff + (uint32_t)buf * (64 * 40 * 2),
              xhAddr + ((uint64_t)pBase * TWOC + (uint64_t)k) * 2,
              /*tile*/ 32, 64, /*tensor*/ TWOC, PIX, /*stride*/ TWOC);
#pragma unroll
    for (int g = 0; g < 3; ++g)
      tdm_issue(bldsOff + (uint32_t)(buf * 3 + g) * (128 * 40 * 2),
                wtAddr + ((uint64_t)g * (CC * TWOC) +
                          (uint64_t)nBase * TWOC + (uint64_t)k) * 2,
                /*tile*/ 32, 128, /*tensor*/ TWOC, CC, /*stride*/ TWOC);
  };

  if (wv == 0) issue(0, 0);

  for (int k0 = 0; k0 < TWOC; k0 += 32) {
    const int cur = (k0 >> 5) & 1;
    if (k0 + 32 < TWOC) {
      if (wv == 0) issue(cur ^ 1, k0 + 32);     // prefetch next K-step
      __builtin_amdgcn_s_wait_tensorcnt(4);     // in-order: prev batch done
    } else {
      __builtin_amdgcn_s_wait_tensorcnt(0);
    }
    __syncthreads();

    // A fragment (16-bit A 16x32 layout): two aligned b128 LDS loads
    v8h a0 = *(const v8h*)&Alds[cur][wm * 16 + ml][hi * 8];
    v8h a1 = *(const v8h*)&Alds[cur][wm * 16 + ml][16 + hi * 8];
    v16h afrag;
#pragma unroll
    for (int e = 0; e < 8; ++e) { afrag[e] = a0[e]; afrag[e + 8] = a1[e]; }

#pragma unroll
    for (int g = 0; g < 3; ++g) {
#pragma unroll
      for (int t = 0; t < 4; ++t) {
        const int n = wn * 64 + t * 16 + ml;
        v8h b0 = *(const v8h*)&Blds[cur][g][n][hi * 8];
        v8h b1 = *(const v8h*)&Blds[cur][g][n][16 + hi * 8];
        v16h bfrag;
#pragma unroll
        for (int e = 0; e < 8; ++e) { bfrag[e] = b0[e]; bfrag[e + 8] = b1[e]; }
        acc[g][t] = __builtin_amdgcn_wmma_f32_16x16x32_f16(
            false, afrag, false, bfrag, (short)0, acc[g][t], false, false);
      }
    }
    __syncthreads();   // all frag reads done before TDM overwrites this buf
  }

  // ---- epilogue: fused activations, pack {fg, delta*ig} as 2x f16 ------
  const int mh = hi * 8;   // C/D layout: lanes 16-31 hold M=8..15
#pragma unroll
  for (int t = 0; t < 4; ++t) {
    const int n   = nBase + wn * 64 + t * 16 + ml;
    const float bii = bi[n];
    const float bff = bf[n];
    const float bdd = bd[n];
#pragma unroll
    for (int r = 0; r < 8; ++r) {
      const int p  = pBase + wm * 16 + mh + r;
      const float ci = acc[0][t][r] + bii;
      const float cf = acc[1][t][r] + bff + 1.0f;
      const float cd = acc[2][t][r] + bdd;
      const float ig = 1.0f / (1.0f + __expf(-ci));
      const float fg = 1.0f / (1.0f + __expf(-cf));
      const float sp = (cd > 20.0f) ? cd : log1pf(__expf(cd));
      const float di = ig * sp;
      union { _Float16 h[2]; unsigned int u; } pk;
      pk.h[0] = (_Float16)fg;
      pk.h[1] = (_Float16)di;
      gates[(size_t)p * CC + n] = pk.u;
    }
  }
}

// ---------------------------------------------------------------------------
// Pass 2: one thread per (h,w,c); fields via 7 complex FMAs against G;
// inner loop over the 4 batch elements. idx2 = (h*128 + w)*256 + c.
// ---------------------------------------------------------------------------
__global__ __launch_bounds__(256) void ssm_final(
    const float* __restrict__ x_re, const float* __restrict__ x_im,
    const unsigned int* __restrict__ gates,
    const float2* __restrict__ G,
    float2* __restrict__ out)
{
  const unsigned idx2 = blockIdx.x * 256u + threadIdx.x;   // < HWCN (4.19M)
  const unsigned wc = idx2 & 32767u;          // (w*256 + c)
  const unsigned h  = idx2 >> 15;             // 0..127

  // h-twiddles r_i = cis(-2*pi*h*(60+i)/128) via 1 sincos pair + recurrence.
  const float STEP = -0.04908738521234052f;  // -2*pi/128
  float2 r[7];
  {
    const int p0 = (int)((h * 60u) & 127u);
    const int ps = (int)(h & 127u);
    float s0, c0, ss, cs;
    __sincosf((float)p0 * STEP, &s0, &c0);
    __sincosf((float)ps * STEP, &ss, &cs);
    r[0] = make_float2(c0, s0);
    const float2 st = make_float2(cs, ss);
#pragma unroll
    for (int i = 1; i < 7; ++i) r[i] = cmulf(r[i - 1], st);
  }

  // Fields: AF/BF/CF = sum_i r_i * G[g][i][w][c]  (21 coalesced b64 loads, L2)
  float2 AF = make_float2(0.f, 0.f);
  float2 BF = make_float2(0.f, 0.f);
  float2 CF = make_float2(0.f, 0.f);
#pragma unroll
  for (int i = 0; i < 7; ++i) {
    const float2 ga = G[(size_t)(0 * 7 + i) * 32768 + wc];
    const float2 gb = G[(size_t)(1 * 7 + i) * 32768 + wc];
    const float2 gc = G[(size_t)(2 * 7 + i) * 32768 + wc];
    AF.x += r[i].x * ga.x - r[i].y * ga.y;  AF.y += r[i].x * ga.y + r[i].y * ga.x;
    BF.x += r[i].x * gb.x - r[i].y * gb.y;  BF.y += r[i].x * gb.y + r[i].y * gb.x;
    CF.x += r[i].x * gc.x - r[i].y * gc.y;  CF.y += r[i].x * gc.y + r[i].y * gc.x;
  }

#pragma unroll
  for (int b = 0; b < BB; ++b) {
    const size_t idx = (size_t)b * HWCN + idx2;

    union { unsigned int u; _Float16 hh[2]; } pk;
    pk.u = gates[idx];
    const float fg = (float)pk.hh[0];
    const float di = (float)pk.hh[1];
    const float xr = x_re[idx];
    const float xi = x_im[idx];

    const float2 d  = make_float2(AF.x * fg, AF.y * fg);
    const float2 bx = cmulf(BF, make_float2(xr, xi));
    const float2 ic = make_float2(di * bx.x, di * bx.y);

    // sum_{t=0..7} d^t = (1+d)(1+d^2)(1+d^4)
    const float2 d2 = cmulf(d, d);
    const float2 d4 = cmulf(d2, d2);
    const float2 s1 = make_float2(1.f + d.x,  d.y);
    const float2 s2 = make_float2(1.f + d2.x, d2.y);
    const float2 s4 = make_float2(1.f + d4.x, d4.y);
    const float2 S  = cmulf(cmulf(s1, s2), s4);
    const float2 hf = cmulf(ic, S);

    out[idx] = cmulf(CF, hf);
  }
}

// ---------------------------------------------------------------------------
extern "C" void kernel_launch(void* const* d_in, const int* in_sizes, int n_in,
                              void* d_out, int out_size, void* d_ws, size_t ws_size,
                              hipStream_t stream) {
  const float* x_re = (const float*)d_in[0];
  const float* x_im = (const float*)d_in[1];
  const float* Wi   = (const float*)d_in[2];
  const float* bi   = (const float*)d_in[3];
  const float* Wf   = (const float*)d_in[4];
  const float* bf   = (const float*)d_in[5];
  const float* Wd   = (const float*)d_in[6];
  const float* bd   = (const float*)d_in[7];
  const float* Ak   = (const float*)d_in[8];
  const float* Bk   = (const float*)d_in[9];
  const float* Ck   = (const float*)d_in[10];

  // workspace layout
  char* ws = (char*)d_ws;
  size_t off = 0;
  unsigned int* gates = (unsigned int*)(ws + off); off += (size_t)PIX * CC * 4;   // 64 MB
  _Float16*     xh    = (_Float16*)(ws + off);     off += (size_t)PIX * TWOC * 2; // 64 MB
  _Float16*     Wt    = (_Float16*)(ws + off);     off += (size_t)3 * CC * TWOC * 2; // 768 KB
  float2*       G     = (float2*)(ws + off);       // 3*7*128*256*8 = 5.5 MB

  convert_w<<<(3 * 16384) / 256, 256, 0, stream>>>(Wi, Wf, Wd, Wt);
  convert_x<<<(PIX * 32) / 256, 256, 0, stream>>>(x_re, x_im, xh);
  build_rowfft<<<(3 * 7 * 128 * 256) / 256, 256, 0, stream>>>(Ak, Bk, Ck, G);

  dim3 gemm_grid(PIX / 64, CC / 128);   // (1024, 2)
  gates_gemm_wmma<<<gemm_grid, 256, 0, stream>>>(xh, Wt, bi, bf, bd, gates);

  ssm_final<<<(HH * WWD * CC) / 256, 256, 0, stream>>>(
      x_re, x_im, gates, G, (float2*)d_out);
}
